// AdaptiveDirectionGenerator_48309792146077
// MI455X (gfx1250) — compile-verified, matrix operand order checked by reading, not run on hardware
//
#include <hip/hip_runtime.h>
#include <hip/hip_bf16.h>

#define DEV __device__ __forceinline__

typedef __attribute__((ext_vector_type(16))) __bf16 v16bf;
typedef __attribute__((ext_vector_type(8)))  float  v8f;
typedef __attribute__((ext_vector_type(4)))  float  v4f;

union Frag {
  v16bf b;
  struct { uint4 lo, hi; } q;
};

DEV v8f wmma_bf16(v16bf a, v16bf b, v8f c) {
  // 8 args: (neg_a, A, neg_b, B, c_mod, C, reuse_a, reuse_b)
  return __builtin_amdgcn_wmma_f32_16x16x32_bf16(false, a, false, b, (short)0, c,
                                                 false, false);
}

// B fragment (32x16 bf16): packed weights P laid out column-major [n*K + k].
// Lanes 0-15 hold column n0+lane, K chunks {k0..k0+7, k0+16..k0+23};
// lanes 16-31 hold the +8 chunks. Two contiguous 16B loads per lane.
DEV v16bf load_bfrag(const __bf16* __restrict__ P, int K, int n0, int k0,
                     int lane) {
  const int half = lane >> 4;
  const __bf16* col = P + (size_t)(n0 + (lane & 15)) * K + k0 + half * 8;
  Frag f;
  f.q.lo = *(const uint4*)(col);
  f.q.hi = *(const uint4*)(col + 16);
  return f.b;
}

// A fragment (16x32 bf16) from row-major bf16 activations H [row*ldk + k].
DEV v16bf load_afrag_bf(const __bf16* __restrict__ H, int ldk, int row0,
                        int k0, int lane) {
  const int half = lane >> 4;
  const __bf16* p = H + (size_t)(row0 + (lane & 15)) * ldk + k0 + half * 8;
  Frag f;
  f.q.lo = *(const uint4*)(p);
  f.q.hi = *(const uint4*)(p + 16);
  return f.b;
}

// A fragment from f32 activations: non-temporal loads (single-use stream,
// keep it out of L2) + native bf16 converts.
DEV v16bf load_afrag_f32(const float* __restrict__ X, int ldk, int row0, int k0,
                         int lane) {
  const int half = lane >> 4;
  const float* p = X + (size_t)(row0 + (lane & 15)) * ldk + k0 + half * 8;
  v4f a0 = __builtin_nontemporal_load((const v4f*)(p));
  v4f a1 = __builtin_nontemporal_load((const v4f*)(p + 4));
  v4f a2 = __builtin_nontemporal_load((const v4f*)(p + 16));
  v4f a3 = __builtin_nontemporal_load((const v4f*)(p + 20));
  Frag f;
  f.b[0]  = (__bf16)a0.x; f.b[1]  = (__bf16)a0.y; f.b[2]  = (__bf16)a0.z; f.b[3]  = (__bf16)a0.w;
  f.b[4]  = (__bf16)a1.x; f.b[5]  = (__bf16)a1.y; f.b[6]  = (__bf16)a1.z; f.b[7]  = (__bf16)a1.w;
  f.b[8]  = (__bf16)a2.x; f.b[9]  = (__bf16)a2.y; f.b[10] = (__bf16)a2.z; f.b[11] = (__bf16)a2.w;
  f.b[12] = (__bf16)a3.x; f.b[13] = (__bf16)a3.y; f.b[14] = (__bf16)a3.z; f.b[15] = (__bf16)a3.w;
  return f.b;
}

DEV float wave_sum(float v) {
  #pragma unroll
  for (int off = 16; off; off >>= 1) v += __shfl_xor(v, off, 32);
  return v;
}
DEV float wave_max(float v) {
  #pragma unroll
  for (int off = 16; off; off >>= 1) v = fmaxf(v, __shfl_xor(v, off, 32));
  return v;
}

// ---------------- weight pack: f32 KxN row-major -> bf16 NxK column-major ----
__global__ __launch_bounds__(256) void k_pack(const float* __restrict__ W,
                                              __bf16* __restrict__ P,
                                              int K, int N) {
  int idx = blockIdx.x * 256 + threadIdx.x;
  if (idx >= K * N) return;
  int n = idx / K, k = idx - n * K;
  P[idx] = (__bf16)W[(size_t)k * N + n];
}

// ------ layer 1: h1 = relu(LN(x @ W1 + b1)); 64 rows/block, 16 waves --------
__global__ __launch_bounds__(512) void k_layer1(const float* __restrict__ X,
    const __bf16* __restrict__ W1p, const float* __restrict__ b1,
    const float* __restrict__ g1, const float* __restrict__ be1,
    __bf16* __restrict__ H1) {
  extern __shared__ float tile[];             // 64 x 512 f32 = 128 KB
  const int lane = threadIdx.x & 31;
  const int wave = threadIdx.x >> 5;          // 0..15
  const int half = lane >> 4, col = lane & 15;
  const int row0 = blockIdx.x * 64;

  v8f acc[4][2] = {};                         // [m-tile][n-tile]
  for (int ks = 0; ks < 768; ks += 32) {
    v16bf a[4];
    #pragma unroll
    for (int m = 0; m < 4; ++m) a[m] = load_afrag_f32(X, 768, row0 + m * 16, ks, lane);
    #pragma unroll
    for (int t = 0; t < 2; ++t) {
      v16bf bf = load_bfrag(W1p, 768, (wave * 2 + t) * 16, ks, lane);
      #pragma unroll
      for (int m = 0; m < 4; ++m) acc[m][t] = wmma_bf16(a[m], bf, acc[m][t]);
    }
  }
  #pragma unroll
  for (int t = 0; t < 2; ++t) {
    const int n0 = (wave * 2 + t) * 16;
    const float bias = b1[n0 + col];
    #pragma unroll
    for (int m = 0; m < 4; ++m)
      #pragma unroll
      for (int r = 0; r < 8; ++r)
        tile[(m * 16 + r + half * 8) * 512 + n0 + col] = acc[m][t][r] + bias;
  }
  __syncthreads();

  for (int rr = 0; rr < 4; ++rr) {
    const int row = wave * 4 + rr;
    float vals[16], s = 0.f, s2 = 0.f;
    #pragma unroll
    for (int t = 0; t < 16; ++t) {
      float v = tile[row * 512 + lane + 32 * t];
      vals[t] = v; s += v; s2 += v * v;
    }
    s = wave_sum(s); s2 = wave_sum(s2);
    const float mu = s * (1.0f / 512.0f);
    const float rstd = rsqrtf(s2 * (1.0f / 512.0f) - mu * mu + 1e-5f);
    #pragma unroll
    for (int t = 0; t < 16; ++t) {
      const int j = lane + 32 * t;
      float v = (vals[t] - mu) * rstd * g1[j] + be1[j];
      H1[(size_t)(row0 + row) * 512 + j] = (__bf16)fmaxf(v, 0.0f);
    }
  }
}

// ------ layer 2: h2 = relu(LN(h1 @ W2 + b2)); 64 rows/block -----------------
__global__ __launch_bounds__(512) void k_layer2(const __bf16* __restrict__ H1,
    const __bf16* __restrict__ W2p, const float* __restrict__ b2,
    const float* __restrict__ g2, const float* __restrict__ be2,
    __bf16* __restrict__ H2) {
  extern __shared__ float tile[];             // 64 x 256 f32 = 64 KB
  const int lane = threadIdx.x & 31;
  const int wave = threadIdx.x >> 5;
  const int half = lane >> 4, col = lane & 15;
  const int row0 = blockIdx.x * 64;

  v8f acc[4] = {};                            // [m-tile], 1 n-tile per wave
  const int n0 = wave * 16;
  for (int ks = 0; ks < 512; ks += 32) {
    v16bf a[4];
    #pragma unroll
    for (int m = 0; m < 4; ++m) a[m] = load_afrag_bf(H1, 512, row0 + m * 16, ks, lane);
    v16bf bf = load_bfrag(W2p, 512, n0, ks, lane);
    #pragma unroll
    for (int m = 0; m < 4; ++m) acc[m] = wmma_bf16(a[m], bf, acc[m]);
  }
  {
    const float bias = b2[n0 + col];
    #pragma unroll
    for (int m = 0; m < 4; ++m)
      #pragma unroll
      for (int r = 0; r < 8; ++r)
        tile[(m * 16 + r + half * 8) * 256 + n0 + col] = acc[m][r] + bias;
  }
  __syncthreads();

  for (int rr = 0; rr < 4; ++rr) {
    const int row = wave * 4 + rr;
    float vals[8], s = 0.f, s2 = 0.f;
    #pragma unroll
    for (int t = 0; t < 8; ++t) {
      float v = tile[row * 256 + lane + 32 * t];
      vals[t] = v; s += v; s2 += v * v;
    }
    s = wave_sum(s); s2 = wave_sum(s2);
    const float mu = s * (1.0f / 256.0f);
    const float rstd = rsqrtf(s2 * (1.0f / 256.0f) - mu * mu + 1e-5f);
    #pragma unroll
    for (int t = 0; t < 8; ++t) {
      const int j = lane + 32 * t;
      float v = (vals[t] - mu) * rstd * g2[j] + be2[j];
      H2[(size_t)(row0 + row) * 256 + j] = (__bf16)fmaxf(v, 0.0f);
    }
  }
}

// ------ attention: attn = softmax(h2 @ Wa + ba); 64 rows/block --------------
__global__ __launch_bounds__(512) void k_attn(const __bf16* __restrict__ H2,
    const __bf16* __restrict__ Wap, const float* __restrict__ ba,
    float* __restrict__ ATT) {
  extern __shared__ float tile[];             // 64 x 768 f32 = 192 KB
  const int lane = threadIdx.x & 31;
  const int wave = threadIdx.x >> 5;
  const int half = lane >> 4, col = lane & 15;
  const int row0 = blockIdx.x * 64;

  v8f acc[4][3] = {};                         // [m-tile][n-tile]
  for (int ks = 0; ks < 256; ks += 32) {
    v16bf a[4];
    #pragma unroll
    for (int m = 0; m < 4; ++m) a[m] = load_afrag_bf(H2, 256, row0 + m * 16, ks, lane);
    #pragma unroll
    for (int t = 0; t < 3; ++t) {
      v16bf bf = load_bfrag(Wap, 256, (wave * 3 + t) * 16, ks, lane);
      #pragma unroll
      for (int m = 0; m < 4; ++m) acc[m][t] = wmma_bf16(a[m], bf, acc[m][t]);
    }
  }
  #pragma unroll
  for (int t = 0; t < 3; ++t) {
    const int n0 = (wave * 3 + t) * 16;
    const float bias = ba[n0 + col];
    #pragma unroll
    for (int m = 0; m < 4; ++m)
      #pragma unroll
      for (int r = 0; r < 8; ++r)
        tile[(m * 16 + r + half * 8) * 768 + n0 + col] = acc[m][t][r] + bias;
  }
  __syncthreads();

  for (int rr = 0; rr < 4; ++rr) {
    const int row = wave * 4 + rr;
    float vals[24], m = -3.0e38f;
    #pragma unroll
    for (int t = 0; t < 24; ++t) {
      vals[t] = tile[row * 768 + lane + 32 * t];
      m = fmaxf(m, vals[t]);
    }
    m = wave_max(m);
    float s = 0.f;
    #pragma unroll
    for (int t = 0; t < 24; ++t) { vals[t] = __expf(vals[t] - m); s += vals[t]; }
    s = wave_sum(s);
    const float inv = 1.0f / s;
    #pragma unroll
    for (int t = 0; t < 24; ++t)
      ATT[(size_t)(row0 + row) * 768 + lane + 32 * t] = vals[t] * inv;
  }
}

// -- directions: W = (h2 @ Wd + bd) * attn ; per-row Gram-Schmidt over K=4 ---
// 32 rows/block: two 16-row M-tiles accumulated in ONE pass over Wd (halves
// the dominant weight traffic), then two stage->scale->GS passes reusing the
// same 192 KB LDS tile. GS stays fully f32. Output stores are non-temporal
// (write-once stream, keep L2 for weights/activations).
__global__ __launch_bounds__(512) void k_dirs(const __bf16* __restrict__ H2,
    const __bf16* __restrict__ Wdp, const float* __restrict__ bd,
    const float* __restrict__ ATT, float* __restrict__ OUT) {
  extern __shared__ float wtile[];            // 16 x 3072 f32 = 192 KB
  const int lane = threadIdx.x & 31;
  const int wave = threadIdx.x >> 5;          // 0..15
  const int half = lane >> 4, col = lane & 15;
  const int row0 = blockIdx.x * 32;

  v8f acc[2][12] = {};                        // [m-tile][n-tile]
  for (int ks = 0; ks < 256; ks += 32) {
    v16bf a0 = load_afrag_bf(H2, 256, row0, ks, lane);
    v16bf a1 = load_afrag_bf(H2, 256, row0 + 16, ks, lane);
    #pragma unroll
    for (int t = 0; t < 12; ++t) {
      v16bf bf = load_bfrag(Wdp, 256, (wave * 12 + t) * 16, ks, lane);
      acc[0][t] = wmma_bf16(a0, bf, acc[0][t]);
      acc[1][t] = wmma_bf16(a1, bf, acc[1][t]);
    }
  }

  #pragma unroll
  for (int mp = 0; mp < 2; ++mp) {
    const int rbase = row0 + mp * 16;
    // stage D + bias into LDS
    #pragma unroll
    for (int t = 0; t < 12; ++t) {
      const int n0 = (wave * 12 + t) * 16;
      const float bias = bd[n0 + col];
      #pragma unroll
      for (int r = 0; r < 8; ++r)
        wtile[(r + half * 8) * 3072 + n0 + col] = acc[mp][t][r] + bias;
    }
    __syncthreads();

    // elementwise attention scaling: W[b,i,j] *= attn[b,j]
    for (int idx = threadIdx.x; idx < 16 * 3072; idx += 512) {
      const int row = idx / 3072;
      const int c = idx - row * 3072;
      wtile[idx] *= ATT[(size_t)(rbase + row) * 768 + (c % 768)];
    }
    __syncthreads();

    // Gram-Schmidt: one row per wave, stride-32 LDS access (conflict-free)
    float* rp = wtile + wave * 3072;
    float v[24];
    for (int i = 0; i < 4; ++i) {
      #pragma unroll
      for (int t = 0; t < 24; ++t) v[t] = rp[i * 768 + lane + 32 * t];
      for (int j = 0; j < i; ++j) {
        float p = 0.f;
        #pragma unroll
        for (int t = 0; t < 24; ++t) p += v[t] * rp[j * 768 + lane + 32 * t];
        p = wave_sum(p);
        #pragma unroll
        for (int t = 0; t < 24; ++t) v[t] -= p * rp[j * 768 + lane + 32 * t];
      }
      float nn = 0.f;
      #pragma unroll
      for (int t = 0; t < 24; ++t) nn += v[t] * v[t];
      nn = wave_sum(nn);
      const float inv = 1.0f / fmaxf(sqrtf(nn), 1e-12f);
      #pragma unroll
      for (int t = 0; t < 24; ++t) {
        const float u = v[t] * inv;
        rp[i * 768 + lane + 32 * t] = u;
        __builtin_nontemporal_store(u,
            &OUT[((size_t)(rbase + wave) * 4 + i) * 768 + lane + 32 * t]);
      }
    }
    __syncthreads();   // GS of this pass must finish before LDS is overwritten
  }
}

extern "C" void kernel_launch(void* const* d_in, const int* in_sizes, int n_in,
                              void* d_out, int out_size, void* d_ws, size_t ws_size,
                              hipStream_t stream) {
  (void)in_sizes; (void)n_in; (void)out_size; (void)ws_size;
  const float* X   = (const float*)d_in[0];
  const float* W1  = (const float*)d_in[1];
  const float* b1  = (const float*)d_in[2];
  const float* g1  = (const float*)d_in[3];
  const float* be1 = (const float*)d_in[4];
  const float* W2  = (const float*)d_in[5];
  const float* b2  = (const float*)d_in[6];
  const float* g2  = (const float*)d_in[7];
  const float* be2 = (const float*)d_in[8];
  const float* Wd  = (const float*)d_in[9];
  const float* bd  = (const float*)d_in[10];
  const float* Wa  = (const float*)d_in[11];
  const float* ba  = (const float*)d_in[12];

  const int B = 16384;
  char* ws = (char*)d_ws;
  size_t off = 0;
  auto carve = [&](size_t bytes) {
    char* p = ws + off;
    off += (bytes + 255) & ~(size_t)255;
    return p;
  };
  __bf16* W1p = (__bf16*)carve((size_t)768 * 512 * 2);
  __bf16* W2p = (__bf16*)carve((size_t)512 * 256 * 2);
  __bf16* Wdp = (__bf16*)carve((size_t)256 * 3072 * 2);
  __bf16* Wap = (__bf16*)carve((size_t)256 * 768 * 2);
  __bf16* H1  = (__bf16*)carve((size_t)B * 512 * 2);
  __bf16* H2  = (__bf16*)carve((size_t)B * 256 * 2);

  float* OUT = (float*)d_out;
  float* ATT = OUT + (size_t)B * 4 * 768;   // outputs: W_ortho then attn

  k_pack<<<(768 * 512 + 255) / 256, 256, 0, stream>>>(W1, W1p, 768, 512);
  k_pack<<<(512 * 256 + 255) / 256, 256, 0, stream>>>(W2, W2p, 512, 256);
  k_pack<<<(256 * 3072 + 255) / 256, 256, 0, stream>>>(Wd, Wdp, 256, 3072);
  k_pack<<<(256 * 768 + 255) / 256, 256, 0, stream>>>(Wa, Wap, 256, 768);

  k_layer1<<<B / 64, 512, 64 * 512 * sizeof(float), stream>>>(X, W1p, b1, g1, be1, H1);
  k_layer2<<<B / 64, 512, 64 * 256 * sizeof(float), stream>>>(H1, W2p, b2, g2, be2, H2);
  k_attn  <<<B / 64, 512, 64 * 768 * sizeof(float), stream>>>(H2, Wap, ba, ATT);
  k_dirs  <<<B / 32, 512, 16 * 3072 * sizeof(float), stream>>>(H2, Wdp, bd, ATT, OUT);
}
